// feedback_20607253086487
// MI455X (gfx1250) — matrix operand order repaired and checked by reading, not compiled, not run on hardware
//
#include <hip/hip_runtime.h>

typedef __attribute__((ext_vector_type(16))) __bf16 v16bf;
typedef __attribute__((ext_vector_type(8)))  __bf16 v8bf;
typedef __attribute__((ext_vector_type(8)))  float  v8f;

#define UNITS   512
#define THREEU  1536
#define TILE_B  32    // batch rows per block (2 WMMA M-tiles -> B fragment reused 2x)
#define HSB     520   // bf16 h row stride (1040B = 65*16 -> 16B-aligned rows, bank-spread)
#define TSTEPS  64
#define OSTEPS  64
#define FDIM    4

union AFrag { v16bf v; v8bf h[2]; };

__device__ __forceinline__ unsigned short f2bf(float f) {
    unsigned u = __float_as_uint(f);
    unsigned r = u + 0x7FFFu + ((u >> 16) & 1u);   // round-to-nearest-even
    return (unsigned short)(r >> 16);
}
__device__ __forceinline__ float bf2f(unsigned short s) {
    return __uint_as_float(((unsigned)s) << 16);
}
__device__ __forceinline__ float sigm(float x)     { return 1.0f / (1.0f + __expf(-x)); }
__device__ __forceinline__ float tanhfast(float x) { return 2.0f / (1.0f + __expf(-2.0f * x)) - 1.0f; }

// Repack rec_kernel [512,1536] fp32 row-major -> bf16 WMMA B-fragment order:
// bp[((nt*16 + kt)*32 + lane)*16 + e], n = nt*16 + (lane&15),
// k = kt*32 + ((lane>=16)?16:0) + e   (contiguous K per lane half, same order as the A loads: lanes 0-15 K 0..15, lanes 16-31 K 16..31)
__global__ void repack_recw_bf16(const float* __restrict__ rec, unsigned short* __restrict__ bp) {
    int d    = blockIdx.x * 256 + threadIdx.x;     // 0 .. 786431
    int e    = d & 15;
    int lane = (d >> 4) & 31;
    int kt   = (d >> 9) & 15;
    int nt   = d >> 13;                            // 0 .. 95
    int n    = nt * 16 + (lane & 15);
    int k    = kt * 32 + ((lane >> 4) & 1) * 16 + e;
    bp[d] = f2bf(rec[k * THREEU + n]);
}

__launch_bounds__(512, 1)
__global__ void gru_forecast(const float* __restrict__ inputs,
                             const float* __restrict__ Kw,
                             const float* __restrict__ bias_in,
                             const float* __restrict__ bias_rec,
                             const unsigned short* __restrict__ Bpack,
                             const float* __restrict__ dense_w,
                             const float* __restrict__ dense_b,
                             float* __restrict__ out) {
    __shared__ unsigned short hb[TILE_B * HSB];    // bf16 hidden state (WMMA A source + dense)
    __shared__ float          xt[TILE_B * FDIM];   // current input / fed-back prediction
    __shared__ float          pp[512];             // dense partial sums

    const int tid  = threadIdx.x;
    const int w    = tid >> 5;                     // 16 waves
    const int lane = tid & 31;
    const int b0   = blockIdx.x * TILE_B;

    for (int i = tid; i < TILE_B * HSB; i += 512) hb[i] = 0;
    __syncthreads();

    const int mA   = lane & 15;            // A-matrix row within tile for this lane
    const int koff = (lane >> 4) << 4;     // 0 / 16 : A K sub-offset per lane half
    const int m2b  = (lane >> 4) << 3;     // 0 / 8 : C-matrix row offset per lane half
    const int ucol = lane & 15;            // C-matrix column within tile
    const unsigned short* blane = Bpack + lane * 16;
    const unsigned short* hrow0 = hb + mA * HSB;          // M-tile 0 (rows 0..15)
    const unsigned short* hrow1 = hb + (16 + mA) * HSB;   // M-tile 1 (rows 16..31)

    // fp32 hidden state lives in registers: element (g,j,r) <-> (m = 16g+m2b+r, u = 32w+16j+ucol)
    float hreg[2][2][8];
    #pragma unroll
    for (int g = 0; g < 2; ++g)
        #pragma unroll
        for (int j = 0; j < 2; ++j)
            #pragma unroll
            for (int r = 0; r < 8; ++r) hreg[g][j][r] = 0.0f;

    for (int step = 0; step < 127; ++step) {
        if (step < TSTEPS) {               // warmup: x from inputs[b, t, :]
            if (tid < TILE_B * FDIM) {
                int m = tid >> 2, f = tid & 3;
                xt[tid] = inputs[(size_t)(b0 + m) * (TSTEPS * FDIM) + step * FDIM + f];
            }
            __syncthreads();
        } // decode: xt holds previous prediction (barrier-protected below)

        // ---- accumulator init: bias_rec (+ bias_in + x@kernel for z/r chunks) ----
        v8f acc[2][6];                     // [m-tile g][chunk c*2 + j]
        #pragma unroll
        for (int c = 0; c < 3; ++c) {
            #pragma unroll
            for (int j = 0; j < 2; ++j) {
                const int u = 32 * w + 16 * j + ucol;
                const int n = c * 512 + u;
                float base = bias_rec[n];
                if (c < 2) base += bias_in[n];
                #pragma unroll
                for (int g = 0; g < 2; ++g) {
                    #pragma unroll
                    for (int r = 0; r < 8; ++r) {
                        float v = base;
                        if (c < 2) {
                            const int m2 = 16 * g + m2b + r;
                            #pragma unroll
                            for (int f = 0; f < 4; ++f) v += xt[m2 * 4 + f] * Kw[f * THREEU + n];
                        }
                        acc[g][c * 2 + j][r] = v;
                    }
                }
            }
        }

        // ---- K loop: mh += h(bf16) @ rec_kernel(bf16); each B fragment feeds 2 WMMAs ----
        #pragma unroll 4
        for (int kt = 0; kt < 16; ++kt) {
            AFrag a0, a1;
            a0.h[0] = *(const v8bf*)(hrow0 + kt * 32 + koff);
            a0.h[1] = *(const v8bf*)(hrow0 + kt * 32 + koff + 8);
            a1.h[0] = *(const v8bf*)(hrow1 + kt * 32 + koff);
            a1.h[1] = *(const v8bf*)(hrow1 + kt * 32 + koff + 8);
            #pragma unroll
            for (int c = 0; c < 3; ++c) {
                #pragma unroll
                for (int j = 0; j < 2; ++j) {
                    const int nt = c * 32 + 2 * w + j;
                    v16bf b = *(const v16bf*)(blane + nt * 8192 + kt * 512);
                    acc[0][c * 2 + j] = __builtin_amdgcn_wmma_f32_16x16x32_bf16(
                        false, a0.v, false, b, (short)0, acc[0][c * 2 + j], false, false);
                    acc[1][c * 2 + j] = __builtin_amdgcn_wmma_f32_16x16x32_bf16(
                        false, a1.v, false, b, (short)0, acc[1][c * 2 + j], false, false);
                }
            }
        }

        // ---- gates + state update (fp32 state stays in registers) ----
        #pragma unroll
        for (int g = 0; g < 2; ++g) {
            #pragma unroll
            for (int j = 0; j < 2; ++j) {
                const int u  = 32 * w + 16 * j + ucol;
                const int nh = 1024 + u;
                const float bih = bias_in[nh];
                #pragma unroll
                for (int r = 0; r < 8; ++r) {
                    const int m2 = 16 * g + m2b + r;
                    float z  = sigm(acc[g][0 + j][r]);
                    float rr = sigm(acc[g][2 + j][r]);
                    float xh = bih;
                    #pragma unroll
                    for (int f = 0; f < 4; ++f) xh += xt[m2 * 4 + f] * Kw[f * THREEU + nh];
                    float cand = tanhfast(xh + rr * acc[g][4 + j][r]);
                    hreg[g][j][r] = z * hreg[g][j][r] + (1.0f - z) * cand;
                }
            }
        }
        __syncthreads();   // all waves done reading old hb
        #pragma unroll
        for (int g = 0; g < 2; ++g) {
            #pragma unroll
            for (int j = 0; j < 2; ++j) {
                const int u = 32 * w + 16 * j + ucol;
                #pragma unroll
                for (int r = 0; r < 8; ++r) {
                    const int m2 = 16 * g + m2b + r;
                    hb[m2 * HSB + u] = f2bf(hreg[g][j][r]);
                }
            }
        }
        __syncthreads();   // new hb visible to all waves / dense head

        // ---- dense head + emission (steps 63..126 -> s = 0..63) ----
        if (step >= 63) {
            const int s = step - 63;
            const int q = tid & 3, f = (tid >> 2) & 3, m = tid >> 4;  // tid = m*16+f*4+q
            float partial = 0.0f;
            const int ub = q * 128;
            for (int u2 = 0; u2 < 128; ++u2)
                partial += bf2f(hb[m * HSB + ub + u2]) * dense_w[(ub + u2) * FDIM + f];
            pp[tid] = partial;
            __syncthreads();
            if (tid < TILE_B * FDIM) {
                const int m3 = tid >> 2, f3 = tid & 3;
                float p = dense_b[f3] + pp[tid * 4] + pp[tid * 4 + 1]
                                      + pp[tid * 4 + 2] + pp[tid * 4 + 3];
                out[((size_t)(b0 + m3) * OSTEPS + s) * FDIM + f3] = p;
                xt[tid] = p;   // feed back as next input (tid = m3*4+f3)
            }
            __syncthreads();   // xt/pp safe before next step reads them
        }
    }
}

extern "C" void kernel_launch(void* const* d_in, const int* in_sizes, int n_in,
                              void* d_out, int out_size, void* d_ws, size_t ws_size,
                              hipStream_t stream) {
    (void)in_sizes; (void)n_in; (void)out_size; (void)ws_size;
    const float* inputs   = (const float*)d_in[0];
    const float* Kw       = (const float*)d_in[1];
    const float* recK     = (const float*)d_in[2];
    const float* bias_in  = (const float*)d_in[3];
    const float* bias_rec = (const float*)d_in[4];
    const float* dense_w  = (const float*)d_in[5];
    const float* dense_b  = (const float*)d_in[6];
    float* out = (float*)d_out;
    unsigned short* Bpack = (unsigned short*)d_ws;   // 512*1536*2 = 1.5 MB

    // One-time (per launch) repack of rec_kernel into WMMA B-fragment order.
    repack_recw_bf16<<<(UNITS * THREEU) / 256, 256, 0, stream>>>(recK, Bpack);
    // Persistent-per-block recurrence: 64 blocks x 32 batch rows, 16 waves each.
    gru_forecast<<<2048 / TILE_B, 512, 0, stream>>>(inputs, Kw, bias_in, bias_rec,
                                                    Bpack, dense_w, dense_b, out);
}
